// PyTorchManualGroupedLinear_36309653520655
// MI455X (gfx1250) — matrix operand order repaired and checked by reading, not compiled, not run on hardware
//
#include <hip/hip_runtime.h>
#include <hip/hip_bf16.h>
#include <stdint.h>

// ---------------- problem constants ----------------
#define NUM_EXPERTS 8
#define IN_F  2048
#define OUT_F 2048
#define TOKENS 32768

// ---------------- tiling ----------------
#define TILE_M 128          // sorted-token rows per block (expert segments padded to this)
#define TILE_N 128          // output columns per block
#define KC     64           // K chunk staged in LDS per stage (2 WMMA k-steps)
#define THREADS 256         // 8 wave32 waves
#define LDS_STRIDE 80       // bf16 elems per LDS row (64 + 16 pad -> 160B, breaks bank conflicts)
#define BUF_BYTES ((TILE_M + TILE_N) * LDS_STRIDE * 2)   // one double-buffer stage (A+B)

// ---------------- workspace layout (uint32 slots) ----------------
#define WS_COUNTS 0         // [8]
#define WS_RAWOFF 8         // [9]  exclusive prefix of counts (+ total)
#define WS_PADOFF 17        // [9]  prefix of counts padded to TILE_M (+ total)
#define WS_CURSOR 26        // [8]  scatter cursors
#define WS_TOKMAP 64        // [TOKENS] token ids grouped by expert

typedef __bf16 bf16x16 __attribute__((ext_vector_type(16)));
typedef float  f32x8   __attribute__((ext_vector_type(8)));

struct FragRaw { uint4 lo, hi; };   // 32 bytes, bit-cast to bf16x16

// fp32 -> bf16 (RNE) via native hardware cvt, packed pair into a dword
static __device__ __forceinline__ unsigned pack_bf16x2(float a, float b) {
  __bf16 x = (__bf16)a;
  __bf16 y = (__bf16)b;
  unsigned short ux = __builtin_bit_cast(unsigned short, x);
  unsigned short uy = __builtin_bit_cast(unsigned short, y);
  return (unsigned)ux | ((unsigned)uy << 16);
}

// ---------------- step 0: clear counters ----------------
__global__ void k_zero(unsigned* ws) {
  int t = threadIdx.x;
  if (t < NUM_EXPERTS) { ws[WS_COUNTS + t] = 0u; ws[WS_CURSOR + t] = 0u; }
}

// ---------------- step 1: histogram ----------------
__global__ void k_hist(const int* __restrict__ assign, unsigned* ws) {
  int t = blockIdx.x * blockDim.x + threadIdx.x;
  if (t < TOKENS) atomicAdd(&ws[WS_COUNTS + assign[t]], 1u);
}

// ---------------- step 2: prefix sums (8 experts, trivial) ----------------
__global__ void k_prefix(unsigned* ws) {
  if (threadIdx.x == 0) {
    unsigned raw = 0, pad = 0;
    for (int e = 0; e < NUM_EXPERTS; ++e) {
      unsigned c = ws[WS_COUNTS + e];
      ws[WS_RAWOFF + e] = raw;
      ws[WS_PADOFF + e] = pad;
      raw += c;
      pad += ((c + TILE_M - 1u) / TILE_M) * TILE_M;
    }
    ws[WS_RAWOFF + NUM_EXPERTS] = raw;
    ws[WS_PADOFF + NUM_EXPERTS] = pad;
  }
}

// ---------------- step 3: scatter token ids by expert ----------------
__global__ void k_scatter(const int* __restrict__ assign, unsigned* ws) {
  int t = blockIdx.x * blockDim.x + threadIdx.x;
  if (t < TOKENS) {
    int e = assign[t];
    unsigned pos = atomicAdd(&ws[WS_CURSOR + e], 1u);
    ws[WS_TOKMAP + ws[WS_RAWOFF + e] + pos] = (unsigned)t;
  }
}

// ---------------- step 4: grouped GEMM, bf16 WMMA, f32 accumulate ----------------
// Double-buffered LDS software pipeline: stage chunk c into buffer (c&1),
// one barrier, compute. Stores at iter c+1 hit the buffer whose readers
// (iter c-1) finished before barrier(c) -> single barrier per chunk.
__global__ __launch_bounds__(THREADS)
void k_gemm(const float* __restrict__ X, const float* __restrict__ W,
            const unsigned* __restrict__ ws, float* __restrict__ Y) {
  __shared__ __align__(16) unsigned char smem[2 * BUF_BYTES + TILE_M * 4];
  int* sTok = (int*)(smem + 2 * BUF_BYTES);

  const unsigned mRow = blockIdx.y * TILE_M;
  if (mRow >= ws[WS_PADOFF + NUM_EXPERTS]) return;   // uniform per block

  // which expert segment does this M tile live in? (segments padded to TILE_M)
  int e = 0;
  while (e < NUM_EXPERTS - 1 && mRow >= ws[WS_PADOFF + e + 1]) ++e;
  const unsigned segStart = ws[WS_RAWOFF + e];
  const unsigned cnt      = ws[WS_COUNTS + e];
  const unsigned rowBase  = mRow - ws[WS_PADOFF + e];

  const int tid = threadIdx.x;
  if (tid < TILE_M) {
    unsigned r = rowBase + (unsigned)tid;
    sTok[tid] = (r < cnt) ? (int)ws[WS_TOKMAP + segStart + r] : -1;
  }
  __syncthreads();

  // ---- cooperative loader assignment: 2 threads per row, 32 cols each ----
  const int ldRow = tid >> 1;
  const int ldCol = (tid & 1) * 32;
  const int tokA  = sTok[ldRow];
  const float* aRow = X + (size_t)(tokA >= 0 ? tokA : 0) * IN_F + ldCol;
  const unsigned oRow = blockIdx.x * TILE_N + (unsigned)ldRow;
  const float* bRow = W + ((size_t)e * OUT_F + oRow) * IN_F + ldCol;
  const int stOffA = ldRow * (LDS_STRIDE * 2) + ldCol * 2;
  const int stOffB = TILE_M * LDS_STRIDE * 2 + stOffA;

  // ---- per-wave compute assignment: 4 waves over M, 2 over N ----
  const int lane = tid & 31;
  const int wid  = tid >> 5;
  const int mB = (wid & 3) * 32;     // 2 x 16-row subtiles
  const int nB = (wid >> 2) * 64;    // 4 x 16-col subtiles
  const int halfSel = lane >> 4;     // upper/lower half-wave
  const int rA = lane & 15;

  f32x8 acc[2][4];
#pragma unroll
  for (int i = 0; i < 2; ++i)
#pragma unroll
    for (int j = 0; j < 4; ++j) acc[i][j] = (f32x8)(0.0f);

  // one pipeline stage: prefetch next chunk, stage current chunk to LDS buffer,
  // barrier, then 2 WMMA k-steps from that buffer
  auto stage = [&](int kc, unsigned char* buf) {
    // prefetch next K chunk into near caches (locality 3 -> WGP scope:
    // ISA prefetch scope 0 pulls into all levels, vs SYS which stops at GL2)
    if (kc + KC < IN_F) {
      __builtin_prefetch(aRow + kc + KC, 0, 3);
      __builtin_prefetch(bRow + kc + KC, 0, 3);
    }
    unsigned char* stA = buf + stOffA;
    unsigned char* stB = buf + stOffB;
    if (tokA >= 0) {
      const float4* p = (const float4*)(aRow + kc);
#pragma unroll
      for (int v = 0; v < 8; ++v) {
        float4 f = p[v];
        uint2 u; u.x = pack_bf16x2(f.x, f.y); u.y = pack_bf16x2(f.z, f.w);
        *(uint2*)(stA + v * 8) = u;
      }
    } else {
      uint2 z; z.x = 0u; z.y = 0u;
#pragma unroll
      for (int v = 0; v < 8; ++v) *(uint2*)(stA + v * 8) = z;
    }
    {
      const float4* p = (const float4*)(bRow + kc);
#pragma unroll
      for (int v = 0; v < 8; ++v) {
        float4 f = p[v];
        uint2 u; u.x = pack_bf16x2(f.x, f.y); u.y = pack_bf16x2(f.z, f.w);
        *(uint2*)(stB + v * 8) = u;
      }
    }
    __syncthreads();

    const uint4* ldsA4 = (const uint4*)buf;
    const uint4* ldsB4 = (const uint4*)(buf + TILE_M * LDS_STRIDE * 2);
#pragma unroll
    for (int ks = 0; ks < 2; ++ks) {        // two 32-deep WMMA steps per chunk
      bf16x16 afr[2]; bf16x16 bfr[4];
      // A 16x32 bf16 fragment: lanes<16 hold K{0..7,16..23}, lanes>=16 K{8..15,24..31}
#pragma unroll
      for (int i = 0; i < 2; ++i) {
        int row = mB + i * 16 + rA;
        int idx = row * (LDS_STRIDE / 8) + ks * 4 + halfSel;
        FragRaw r; r.lo = ldsA4[idx]; r.hi = ldsA4[idx + 2];
        afr[i] = __builtin_bit_cast(bf16x16, r);
      }
      // B 32x16 bf16 fragment: lanes<16 hold K{0..7,16..23}, lanes>=16 K{8..15,24..31}
#pragma unroll
      for (int j = 0; j < 4; ++j) {
        int row = nB + j * 16 + rA;          // weight row o == output column n
        int idx = row * (LDS_STRIDE / 8) + ks * 4 + halfSel;
        FragRaw r; r.lo = ldsB4[idx]; r.hi = ldsB4[idx + 2];
        bfr[j] = __builtin_bit_cast(bf16x16, r);
      }
#pragma unroll
      for (int i = 0; i < 2; ++i)
#pragma unroll
        for (int j = 0; j < 4; ++j)
          acc[i][j] = __builtin_amdgcn_wmma_f32_16x16x32_bf16(
              false, afr[i], false, bfr[j], (short)0, acc[i][j], false, false);
    }
  };

  // K loop unrolled by 2 so each buffer's LDS offsets are immediates
  for (int kc = 0; kc < IN_F; kc += 2 * KC) {
    stage(kc,       smem);
    stage(kc + KC,  smem + BUF_BYTES);
  }

  // ---- scatter-store: C layout VGPR v -> M = v (+8 for upper half-wave), N = lane&15 ----
  const int colBase = blockIdx.x * TILE_N + nB + rA;
#pragma unroll
  for (int i = 0; i < 2; ++i) {
#pragma unroll
    for (int v = 0; v < 8; ++v) {
      int tok = sTok[mB + i * 16 + halfSel * 8 + v];
      if (tok >= 0) {
        float* orow = Y + (size_t)tok * OUT_F + colBase;
#pragma unroll
        for (int j = 0; j < 4; ++j) orow[j * 16] = acc[i][j][v];
      }
    }
  }
}

// ---------------- host-side launch ----------------
extern "C" void kernel_launch(void* const* d_in, const int* in_sizes, int n_in,
                              void* d_out, int out_size, void* d_ws, size_t ws_size,
                              hipStream_t stream) {
  (void)in_sizes; (void)n_in; (void)out_size; (void)ws_size;
  const float* X      = (const float*)d_in[0];
  const float* W      = (const float*)d_in[1];
  const int*   assign = (const int*)d_in[2];
  float*       Y      = (float*)d_out;
  unsigned*    ws     = (unsigned*)d_ws;

  k_zero   <<<1, 32, 0, stream>>>(ws);
  k_hist   <<<TOKENS / 256, 256, 0, stream>>>(assign, ws);
  k_prefix <<<1, 32, 0, stream>>>(ws);
  k_scatter<<<TOKENS / 256, 256, 0, stream>>>(assign, ws);

  dim3 grid(OUT_F / TILE_N, (TOKENS + NUM_EXPERTS * TILE_M) / TILE_M);  // 16 x 264
  k_gemm<<<grid, THREADS, 0, stream>>>(X, W, ws, Y);
}
